// Attention_72627896975608
// MI455X (gfx1250) — compile-verified, matrix operand order checked by reading, not run on hardware
//
#include <hip/hip_runtime.h>
#include <hip/hip_bf16.h>
#include <math.h>

typedef __attribute__((ext_vector_type(16))) _Float16 v16h;
typedef __attribute__((ext_vector_type(8)))  _Float16 v8h;
typedef __attribute__((ext_vector_type(8)))  float    v8f;
typedef __attribute__((ext_vector_type(4)))  float    f32x4;

#define B_  8
#define L_  1024
#define C_  256
#define H_  8
#define CH_ 32

// K-index map for 16-bit A/B WMMA fragments (16x32 / 32x16), per ISA 7.12.2:
// lane = hi*16 + (lane&15); element j covers K = (j&7) + 16*(j>>3) + 8*hi
// -> per lane the fragment is two contiguous 8-half chunks at K = 8*hi and 16+8*hi.
__device__ __forceinline__ v16h combine8(v8h a, v8h b) {
    return __builtin_shufflevector(a, b, 0, 1, 2, 3, 4, 5, 6, 7,
                                         8, 9, 10, 11, 12, 13, 14, 15);
}

__device__ __forceinline__ v16h load_frag16(const _Float16* row, int hi) {
    const v8h* p = (const v8h*)row;
    return combine8(p[hi], p[2 + hi]);
}

// ---------------------------------------------------------------------------
// Kernel 0: convert all 5 weight matrices to f16, transposed to [N][K] so
// B-fragment loads are two contiguous b128 loads per lane.
// ---------------------------------------------------------------------------
__global__ void __launch_bounds__(256)
convert_weights_kernel(const float* __restrict__ Wq, const float* __restrict__ Wk,
                       const float* __restrict__ Wv, const float* __restrict__ Wg,
                       const float* __restrict__ Wo, _Float16* __restrict__ Wt)
{
    const int idx = blockIdx.x * 256 + threadIdx.x;   // element in [K][N]
    const int p   = blockIdx.y;                       // 0..4
    const float* W = (p == 0) ? Wq : (p == 1) ? Wk : (p == 2) ? Wv
                   : (p == 3) ? Wg : Wo;
    const int k = idx >> 8, n = idx & 255;
    Wt[(size_t)p * 65536 + (size_t)n * 256 + k] = (_Float16)W[idx];
}

// ---------------------------------------------------------------------------
// Kernel 1: fused QKV+G projections.  X[8192x256] * W -> 16x64 per wave
// (4 accumulators, A reused 4x).  Writes q (scaled), k, gate in [B,H,L,CH],
// and V TRANSPOSED in [B,H,CH,L] so attention's P*V B-fragments are contiguous.
// All 4 B fragments are loaded before the 4 WMMAs so loads clause + overlap.
// ---------------------------------------------------------------------------
__global__ void __launch_bounds__(32)
proj_qkvg_kernel(const float* __restrict__ X, const _Float16* __restrict__ Wt,
                 const float* __restrict__ bg,
                 _Float16* __restrict__ qh, _Float16* __restrict__ kh,
                 _Float16* __restrict__ vhT, _Float16* __restrict__ gh)
{
    const int lane = threadIdx.x & 31;
    const int m = lane & 15, hi = lane >> 4, n = m;
    const int row0 = blockIdx.x * 16;
    const int col0 = blockIdx.y * 64;
    const int proj = blockIdx.z;

    // hoisted per-tile B row pointers (addresses become base + imm offset)
    const _Float16* wr[4];
#pragma unroll
    for (int t = 0; t < 4; ++t)
        wr[t] = Wt + (size_t)proj * 65536 + (size_t)(col0 + t * 16 + n) * 256;

    v8f acc[4] = {};
    const float* xr = X + (size_t)(row0 + m) * C_;
#pragma unroll
    for (int k0 = 0; k0 < C_; k0 += 32) {
        // A fragment: rows of X, f32 -> f16, two 8-float contiguous chunks
        const f32x4* x0 = (const f32x4*)(xr + k0 + (hi << 3));
        const f32x4* x1 = (const f32x4*)(xr + k0 + 16 + (hi << 3));
        const f32x4 a0 = x0[0], a1 = x0[1], b0 = x1[0], b1 = x1[1];
        v16h a;
#pragma unroll
        for (int j = 0; j < 4; ++j) {
            a[j]      = (_Float16)a0[j];
            a[4 + j]  = (_Float16)a1[j];
            a[8 + j]  = (_Float16)b0[j];
            a[12 + j] = (_Float16)b1[j];
        }
        // load all 4 B fragments first, then issue the 4 WMMAs
        v16h bf[4];
#pragma unroll
        for (int t = 0; t < 4; ++t)
            bf[t] = load_frag16(wr[t] + k0, hi);
#pragma unroll
        for (int t = 0; t < 4; ++t)
            acc[t] = __builtin_amdgcn_wmma_f32_16x16x32_f16(false, a, false, bf[t],
                                                            (short)0, acc[t],
                                                            false, false);
    }

    const float qscale = 0.17677669529663687f;  // 1/sqrt(32)
#pragma unroll
    for (int t = 0; t < 4; ++t) {
        const int col = col0 + t * 16 + n;
        const int h = col >> 5, ch = col & (CH_ - 1);
        const float bgv = (proj == 3) ? bg[col] : 0.f;
#pragma unroll
        for (int r = 0; r < 8; ++r) {
            const int row = row0 + r + (hi << 3);
            const int bb = row >> 10, l = row & (L_ - 1);
            const size_t hdr = (size_t)(bb * H_ + h);
            const float v = acc[t][r];
            if (proj == 0) {
                qh[(hdr * L_ + l) * CH_ + ch] = (_Float16)(v * qscale);
            } else if (proj == 1) {
                kh[(hdr * L_ + l) * CH_ + ch] = (_Float16)v;
            } else if (proj == 2) {
                vhT[(hdr * CH_ + ch) * L_ + l] = (_Float16)v;   // transposed
            } else {
                const float g = 1.0f / (1.0f + __expf(-(v + bgv)));
                gh[(hdr * L_ + l) * CH_ + ch] = (_Float16)g;
            }
        }
    }
}

// ---------------------------------------------------------------------------
// Kernel 2: flash attention with bias add + gating.
// k and v fragments both issued at the top of the key loop so their latency
// hides under the score WMMAs + softmax VALU work.  bias_pair (256 MB,
// read-once) streamed with non-temporal loads + prefetch of the next tile.
// ---------------------------------------------------------------------------
__global__ void __launch_bounds__(256)
attention_kernel(const _Float16* __restrict__ qh, const _Float16* __restrict__ kh,
                 const _Float16* __restrict__ vhT, const _Float16* __restrict__ gh,
                 const float* __restrict__ bias_mask,
                 const float* __restrict__ bias_pair,
                 _Float16* __restrict__ oh)
{
    __shared__ _Float16 lds_p[8][16][32];  // per-wave probability tile (8 KB)

    const int wave = threadIdx.x >> 5;
    const int lane = threadIdx.x & 31;
    const int m = lane & 15, hi = lane >> 4, n = m;
    const int b = blockIdx.z, h = blockIdx.y;
    const int q0 = (blockIdx.x * 8 + wave) * 16;
    const size_t bh = (size_t)(b * H_ + h);

    const _Float16* qbase = qh + (bh * L_ + q0) * CH_;
    const _Float16* kr0b  = kh + (bh * L_ + n) * CH_;        // key row n
    const _Float16* kr1b  = kh + (bh * L_ + 16 + n) * CH_;   // key row 16+n
    const _Float16* vr0b  = vhT + (bh * CH_ + n) * L_;       // ch row n
    const _Float16* vr1b  = vhT + (bh * CH_ + 16 + n) * L_;  // ch row 16+n
    const float* bm = bias_mask + (size_t)b * L_;

    // per-row bias_pair pointers (hoisted)
    const float* bprow[8];
#pragma unroll
    for (int r = 0; r < 8; ++r)
        bprow[r] = bias_pair + bh * L_ * L_
                 + (size_t)(q0 + r + (hi << 3)) * L_ + n;

    // q tile = one 16x32 A-fragment (CH == K == 32): two b128 loads
    const v16h qf = load_frag16(qbase + m * CH_, hi);

    float mi[8], li[8];
#pragma unroll
    for (int r = 0; r < 8; ++r) { mi[r] = -INFINITY; li[r] = 0.f; }
    v8f o0 = {}, o1 = {};   // output accum: ch 0-15 / 16-31

    for (int key0 = 0; key0 < L_; key0 += 32) {
        // ---- issue ALL k and v fragment loads up front (one clause) ----
        const v16h kf0 = load_frag16(kr0b + (size_t)key0 * CH_, hi);
        const v16h kf1 = load_frag16(kr1b + (size_t)key0 * CH_, hi);
        const v16h vf0 = load_frag16(vr0b + key0, hi);
        const v16h vf1 = load_frag16(vr1b + key0, hi);

        v8f zero = {};
        v8f s0 = __builtin_amdgcn_wmma_f32_16x16x32_f16(false, qf, false, kf0,
                                                        (short)0, zero, false, false);
        v8f s1 = __builtin_amdgcn_wmma_f32_16x16x32_f16(false, qf, false, kf1,
                                                        (short)0, zero, false, false);

        // bias add (read-once stream: non-temporal)
        const float bm0 = bm[key0 + n];
        const float bm1 = bm[key0 + 16 + n];
#pragma unroll
        for (int r = 0; r < 8; ++r) {
            s0[r] += __builtin_nontemporal_load(bprow[r] + key0) + bm0;
            s1[r] += __builtin_nontemporal_load(bprow[r] + key0 + 16) + bm1;
        }
        if (key0 + 32 < L_)
            __builtin_prefetch(bprow[hi * 4] + key0 + 32, 0, 0);

        // online softmax: row r+8*hi lives in one 16-lane half -> xor masks 1..8
        v8f p0, p1;
#pragma unroll
        for (int r = 0; r < 8; ++r) {
            float t = fmaxf(s0[r], s1[r]);
#pragma unroll
            for (int off = 8; off >= 1; off >>= 1)
                t = fmaxf(t, __shfl_xor(t, off, 32));
            const float mnew  = fmaxf(mi[r], t);
            const float scale = __expf(mi[r] - mnew);
            const float e0 = __expf(s0[r] - mnew);
            const float e1 = __expf(s1[r] - mnew);
            p0[r] = e0; p1[r] = e1;
            float rs = e0 + e1;
#pragma unroll
            for (int off = 8; off >= 1; off >>= 1)
                rs += __shfl_xor(rs, off, 32);
            li[r] = li[r] * scale + rs;
            mi[r] = mnew;
            o0[r] *= scale;
            o1[r] *= scale;
        }

        // P: D-layout -> LDS -> A-fragment (same-wave, DScnt-ordered)
#pragma unroll
        for (int r = 0; r < 8; ++r) {
            lds_p[wave][r + (hi << 3)][n]      = (_Float16)p0[r];
            lds_p[wave][r + (hi << 3)][16 + n] = (_Float16)p1[r];
        }
        const v16h pf = load_frag16(&lds_p[wave][m][0], hi);

        o0 = __builtin_amdgcn_wmma_f32_16x16x32_f16(false, pf, false, vf0,
                                                    (short)0, o0, false, false);
        o1 = __builtin_amdgcn_wmma_f32_16x16x32_f16(false, pf, false, vf1,
                                                    (short)0, o1, false, false);
    }

    // normalize, gate, store f16 [B, L, H*CH]
#pragma unroll
    for (int r = 0; r < 8; ++r) {
        const int l = q0 + r + (hi << 3);
        const float inv = 1.0f / li[r];
        const float g0 = (float)gh[(bh * L_ + l) * CH_ + n];
        const float g1 = (float)gh[(bh * L_ + l) * CH_ + 16 + n];
        _Float16* orow = oh + ((size_t)b * L_ + l) * C_ + h * CH_;
        orow[n]      = (_Float16)(o0[r] * inv * g0);
        orow[16 + n] = (_Float16)(o1[r] * inv * g1);
    }
}

// ---------------------------------------------------------------------------
// Kernel 3: output projection  oh[8192x256] (f16) * Wo^T (f16) + bo -> f32.
// 16x64 per wave, 4 accumulators, batched b128 fragment loads.
// ---------------------------------------------------------------------------
__global__ void __launch_bounds__(32)
out_proj_kernel(const _Float16* __restrict__ oh, const _Float16* __restrict__ Wt,
                const float* __restrict__ bo, float* __restrict__ out)
{
    const int lane = threadIdx.x & 31;
    const int m = lane & 15, hi = lane >> 4, n = m;
    const int row0 = blockIdx.x * 16;
    const int col0 = blockIdx.y * 64;

    const _Float16* wr[4];
#pragma unroll
    for (int t = 0; t < 4; ++t)
        wr[t] = Wt + (size_t)4 * 65536 + (size_t)(col0 + t * 16 + n) * 256;

    v8f acc[4] = {};
    const _Float16* ar = oh + (size_t)(row0 + m) * C_;
#pragma unroll
    for (int k0 = 0; k0 < C_; k0 += 32) {
        const v16h a = load_frag16(ar + k0, hi);
        v16h bf[4];
#pragma unroll
        for (int t = 0; t < 4; ++t)
            bf[t] = load_frag16(wr[t] + k0, hi);
#pragma unroll
        for (int t = 0; t < 4; ++t)
            acc[t] = __builtin_amdgcn_wmma_f32_16x16x32_f16(false, a, false, bf[t],
                                                            (short)0, acc[t],
                                                            false, false);
    }
#pragma unroll
    for (int t = 0; t < 4; ++t) {
        const int col = col0 + t * 16 + n;
        const float bias = bo[col];
#pragma unroll
        for (int r = 0; r < 8; ++r)
            out[(size_t)(row0 + r + (hi << 3)) * C_ + col] = acc[t][r] + bias;
    }
}

// ---------------------------------------------------------------------------
extern "C" void kernel_launch(void* const* d_in, const int* in_sizes, int n_in,
                              void* d_out, int out_size, void* d_ws, size_t ws_size,
                              hipStream_t stream)
{
    const float* q_x       = (const float*)d_in[0];
    const float* bias_mask = (const float*)d_in[1];
    const float* bias_pair = (const float*)d_in[2];
    const float* Wq        = (const float*)d_in[3];
    const float* Wk        = (const float*)d_in[4];
    const float* Wv        = (const float*)d_in[5];
    const float* Wg        = (const float*)d_in[6];
    const float* bg        = (const float*)d_in[7];
    const float* Wo        = (const float*)d_in[8];
    const float* bo        = (const float*)d_in[9];

    char* ws = (char*)d_ws;
    const size_t SZ = (size_t)B_ * H_ * L_ * CH_ * sizeof(_Float16);  // 4 MB
    _Float16* qh  = (_Float16*)(ws);
    _Float16* kh  = (_Float16*)(ws + SZ);
    _Float16* vhT = (_Float16*)(ws + 2 * SZ);
    _Float16* gh  = (_Float16*)(ws + 3 * SZ);
    _Float16* oh  = (_Float16*)(ws + 4 * SZ);
    _Float16* Wt  = (_Float16*)(ws + 5 * SZ);   // 5 x 128 KB f16 transposed

    // 0) weights -> f16 [N][K]
    convert_weights_kernel<<<dim3(256, 5), 256, 0, stream>>>(
        Wq, Wk, Wv, Wg, Wo, Wt);

    // 1) fused QKV+G projections: grid (M/16, N/64, 4 projections)
    proj_qkvg_kernel<<<dim3(512, 4, 4), 32, 0, stream>>>(
        q_x, Wt, bg, qh, kh, vhT, gh);

    // 2) attention: grid (q-chunks, H, B), 8 waves/block (one 16-q tile each)
    attention_kernel<<<dim3(8, 8, 8), 256, 0, stream>>>(
        qh, kh, vhT, gh, bias_mask, bias_pair, oh);

    // 3) output projection: grid (M/16, N/64)
    out_proj_kernel<<<dim3(512, 4), 32, 0, stream>>>(
        oh, Wt, bo, (float*)d_out);
}
